// VoxelSetAbstractionFFps_51960514347368
// MI455X (gfx1250) — matrix operand order repaired and checked by reading, not compiled, not run on hardware
//
#include <hip/hip_runtime.h>
#include <stdint.h>

// ---------------- config ----------------
#define BATCH   2
#define NPT     16384
#define CF      32
#define NK      2048
#define NK_FS   1364          // NK//3*2
#define NK_DS   684
#define H_BEV   200
#define W_BEV   176
#define C_BEV   256
#define C_OUT   128
#define KPAD    512           // fused feature row padded to 512 (480 real)
#define NS1     16
#define NS2     32
#define R1SQ    (0.8f*0.8f)
#define R2SQ    (1.6f*1.6f)

typedef _Float16 v16h __attribute__((ext_vector_type(16)));
typedef float    v8f  __attribute__((ext_vector_type(8)));

union FragA { v16h h; uint32_t u[8]; };
union FragB { v16h h; uint32_t u[8]; };

static __device__ __forceinline__ v8f vzero8() {
  v8f z = {0.f,0.f,0.f,0.f,0.f,0.f,0.f,0.f};
  return z;
}

// ---------------- workspace layout (bytes) ----------------
#define OFF_KPIDX  ((size_t)0)          // B*NK int            = 16384
#define OFF_KP     ((size_t)16384)      // B*NK*3 f32          = 49152
#define OFF_CNT    ((size_t)65536)      // B*NK*2 int          = 32768
#define OFF_IDX1   ((size_t)98304)      // B*NK*16 int         = 262144
#define OFF_IDX2   ((size_t)360448)     // B*NK*32 int         = 524288
#define OFF_W1A    ((size_t)884736)     // 64*64  f16
#define OFF_W2A    ((size_t)892928)     // 64*64  f16
#define OFF_W1B    ((size_t)901120)     // 64*64  f16
#define OFF_W2B    ((size_t)909312)     // 128*64 f16
#define OFF_FW     ((size_t)925696)     // 128*512 f16
#define OFF_FEAT   ((size_t)1056768)    // B*NK*512 f16        = 4 MB

// ---------------- weight transpose + f16 cast ----------------
// All weights stored N-major [N][Kpad] so WMMA B-fragment dword loads are contiguous.
__global__ void prep_weights_kernel(const float* W1a, const float* W2a,
                                    const float* W1b, const float* W2b,
                                    const float* fW,
                                    _Float16* W1at, _Float16* W2at,
                                    _Float16* W1bt, _Float16* W2bt,
                                    _Float16* fWt) {
  int i = blockIdx.x * blockDim.x + threadIdx.x;
  if (i < 4096) {                       // W1a: (35,64) -> [64][64], K zero-padded
    int n = i >> 6, k = i & 63;
    W1at[i] = (_Float16)((k < 35) ? W1a[k*64 + n] : 0.f);
  } else if (i < 8192) {                // W2a: (64,64) -> [64][64]
    int j = i - 4096; int n = j >> 6, k = j & 63;
    W2at[j] = (_Float16)W2a[k*64 + n];
  } else if (i < 12288) {               // W1b: (35,64) -> [64][64]
    int j = i - 8192; int n = j >> 6, k = j & 63;
    W1bt[j] = (_Float16)((k < 35) ? W1b[k*64 + n] : 0.f);
  } else if (i < 20480) {               // W2b: (64,128) -> [128][64]
    int j = i - 12288; int n = j >> 6, k = j & 63;
    W2bt[j] = (_Float16)W2b[k*128 + n];
  } else if (i < 86016) {               // fuse_W: (480,128) -> [128][512]
    int j = i - 20480; int n = j >> 9, k = j & 511;
    fWt[j] = (_Float16)((k < 480) ? fW[k*128 + n] : 0.f);
  }
}

// ---------------- FPS (sequential argmax chain, one block per (batch,mode)) ----------------
__global__ __launch_bounds__(512) void fps_kernel(const float* pts, const float* feats,
                                                  int* kp_idx) {
  const int mode = blockIdx.x & 1;          // 0 = feature-space, 1 = xyz
  const int b    = blockIdx.x >> 1;
  const int dim  = mode ? 3 : CF;
  const float* src = mode ? (pts + (size_t)b*NPT*3) : (feats + (size_t)b*NPT*CF);
  int* out = kp_idx + b*NK + (mode ? NK_FS : 0);
  const int K = mode ? NK_DS : NK_FS;
  const int tid = threadIdx.x;
  const int PT = NPT / 512;                 // 32 points per thread

  float d[PT];
  #pragma unroll
  for (int p = 0; p < PT; ++p) d[p] = 3.4e38f;

  __shared__ float ref[CF];
  __shared__ float rv[16];
  __shared__ int   ri[16];
  __shared__ int   s_last;
  if (tid == 0) { out[0] = 0; s_last = 0; }
  __syncthreads();
  int last = 0;

  for (int it = 1; it < K; ++it) {
    if (tid < dim) ref[tid] = src[(size_t)last*dim + tid];
    __syncthreads();
    float bv = -1.f; int bi = 0;
    for (int p = 0; p < PT; ++p) {
      int i = p*512 + tid;
      const float* row = src + (size_t)i*dim;
      float s = 0.f;
      for (int c = 0; c < dim; ++c) { float t = row[c] - ref[c]; s += t*t; }
      float dn = fminf(d[p], s);
      d[p] = dn;
      if (dn > bv) { bv = dn; bi = i; }
    }
    // wave32 argmax reduce
    #pragma unroll
    for (int o = 16; o > 0; o >>= 1) {
      float ov = __shfl_xor(bv, o, 32);
      int   oi = __shfl_xor(bi, o, 32);
      if (ov > bv || (ov == bv && oi < bi)) { bv = ov; bi = oi; }
    }
    int w = tid >> 5;
    if ((tid & 31) == 0) { rv[w] = bv; ri[w] = bi; }
    __syncthreads();
    if (tid == 0) {
      float mv = rv[0]; int mi = ri[0];
      for (int k2 = 1; k2 < 16; ++k2)
        if (rv[k2] > mv || (rv[k2] == mv && ri[k2] < mi)) { mv = rv[k2]; mi = ri[k2]; }
      out[it] = mi; s_last = mi;
    }
    __syncthreads();
    last = s_last;
  }
}

// ---------------- gather keypoints + write coords output ----------------
__global__ void gather_kp_kernel(const float* pts, const int* kp_idx,
                                 float* kp, float* coords) {
  int i = blockIdx.x * blockDim.x + threadIdx.x;
  if (i >= BATCH*NK) return;
  int b = i / NK;
  int idx = kp_idx[i];
  const float* p = pts + ((size_t)b*NPT + idx)*3;
  float x = p[0], y = p[1], z = p[2];
  kp[i*3+0] = x; kp[i*3+1] = y; kp[i*3+2] = z;
  coords[i*4+0] = (float)b;
  coords[i*4+1] = x; coords[i*4+2] = y; coords[i*4+3] = z;
}

// ---------------- ball query: one wave32 per keypoint, ordered append via ballot ----------------
__global__ __launch_bounds__(256) void ball_query_kernel(const float* pts, const float* kp,
                                                         int* idx1, int* idx2, int* cnts) {
  int kpi  = blockIdx.x*8 + (threadIdx.x >> 5);
  int lane = threadIdx.x & 31;
  int b = kpi / NK;
  float kx = kp[kpi*3+0], ky = kp[kpi*3+1], kz = kp[kpi*3+2];
  const float* P = pts + (size_t)b*NPT*3;
  int* o1 = idx1 + kpi*NS1;
  int* o2 = idx2 + kpi*NS2;
  int c1 = 0, c2 = 0, first1 = 0, first2 = 0;

  for (int base = 0; base < NPT && (c1 < NS1 || c2 < NS2); base += 32) {
    int i = base + lane;
    float dx = P[i*3+0]-kx, dy = P[i*3+1]-ky, dz = P[i*3+2]-kz;
    float d2v = dx*dx + dy*dy + dz*dz;
    unsigned m2 = __builtin_amdgcn_ballot_w32(d2v <= R2SQ);
    unsigned m1 = __builtin_amdgcn_ballot_w32(d2v <= R1SQ);
    unsigned lt = (lane == 0) ? 0u : (0xFFFFFFFFu >> (32 - lane));
    if (c2 == 0 && m2) first2 = base + __builtin_ctz(m2);
    if (c1 == 0 && m1) first1 = base + __builtin_ctz(m1);
    if (m2 & (1u << lane)) { int pos = c2 + __popc(m2 & lt); if (pos < NS2) o2[pos] = i; }
    if (m1 & (1u << lane)) { int pos = c1 + __popc(m1 & lt); if (pos < NS1) o1[pos] = i; }
    c2 += __popc(m2);
    c1 += __popc(m1);
  }
  c1 = c1 < NS1 ? c1 : NS1;
  c2 = c2 < NS2 ? c2 : NS2;
  if (lane == 0) { cnts[kpi*2+0] = c1; cnts[kpi*2+1] = c2; }
  // replicate-first padding (CUDA ball_query semantics)
  for (int j = lane; j < NS1; j += 32) if (j >= c1) o1[j] = first1;
  for (int j = lane; j < NS2; j += 32) if (j >= c2) o2[j] = first2;
}

// ---------------- grouped MLP + max-pool via WMMA; one wave per keypoint ----------------
// Layer1: [NS x 64pad] @ [64 x 64]  -> relu -> H1 (LDS)
// Layer2: [NS x 64]    @ [64 x C2]  -> relu -> max over NS rows -> feat columns
template<int NS, int C2>
__global__ __launch_bounds__(256) void group_mlp_kernel(
    const float* pts, const float* feats, const float* kp,
    const int* idxbuf, const int* cnts, int cnt_sel,
    const _Float16* __restrict__ W1t, const float* __restrict__ b1,
    const _Float16* __restrict__ W2t, const float* __restrict__ b2,
    _Float16* featbuf, int colofs) {
  constexpr int MT  = NS / 16;
  constexpr int NT2 = C2 / 16;
  __shared__ __align__(16) _Float16 stage_all[8][NS*64];

  int w = threadIdx.x >> 5, lane = threadIdx.x & 31;
  int kpi = blockIdx.x*8 + w;
  int b = kpi / NK;
  _Float16* st = stage_all[w];
  const int* gidx = idxbuf + kpi*NS;
  float kx = kp[kpi*3+0], ky = kp[kpi*3+1], kz = kp[kpi*3+2];
  const float* P = pts   + (size_t)b*NPT*3;
  const float* F = feats + (size_t)b*NPT*CF;

  // stage grouped inputs [NS][64] f16 (rel-xyz | feats | zero pad)
  for (int e = lane; e < NS*64; e += 32) {
    int row = e >> 6, col = e & 63;
    int gi = gidx[row];
    float v;
    if      (col < 3)  v = P[gi*3+col] - (col == 0 ? kx : (col == 1 ? ky : kz));
    else if (col < 35) v = F[gi*CF + col - 3];
    else               v = 0.f;
    st[e] = (_Float16)v;
  }
  __syncthreads();

  int half = lane >> 4, nl = lane & 15;

  // ---- layer 1 ----
  v8f acc1[MT][4];
  #pragma unroll
  for (int mt = 0; mt < MT; ++mt)
    #pragma unroll
    for (int t = 0; t < 4; ++t) acc1[mt][t] = vzero8();

  #pragma unroll
  for (int s = 0; s < 2; ++s) {
    FragA a[MT];
    #pragma unroll
    for (int mt = 0; mt < MT; ++mt) {
      int m = mt*16 + nl;
      #pragma unroll
      for (int v = 0; v < 8; ++v) {
        int k = s*32 + (v >> 2)*16 + half*8 + (v & 3)*2;   // 16-bit A layout
        a[mt].u[v] = *(const uint32_t*)&st[m*64 + k];
      }
    }
    #pragma unroll
    for (int t = 0; t < 4; ++t) {
      FragB bf;
      int n = t*16 + nl;
      #pragma unroll
      for (int v = 0; v < 8; ++v) {
        int k = s*32 + (v >> 2)*16 + half*8 + (v & 3)*2;   // 16-bit B layout (same k order as A)
        bf.u[v] = *(const uint32_t*)&W1t[n*64 + k];
      }
      #pragma unroll
      for (int mt = 0; mt < MT; ++mt)
        acc1[mt][t] = __builtin_amdgcn_wmma_f32_16x16x32_f16(
            false, a[mt].h, false, bf.h, (short)0, acc1[mt][t], false, false);
    }
  }
  __syncthreads();

  // bias + relu, write H1 back into stage
  #pragma unroll
  for (int mt = 0; mt < MT; ++mt)
    #pragma unroll
    for (int t = 0; t < 4; ++t) {
      int n = t*16 + nl;
      float bb = b1[n];
      #pragma unroll
      for (int r = 0; r < 8; ++r) {
        int m = mt*16 + half*8 + r;                        // C layout: M = r + half*8
        st[m*64 + n] = (_Float16)fmaxf(acc1[mt][t][r] + bb, 0.f);
      }
    }
  __syncthreads();

  // ---- layer 2 ----
  FragA a2[2][MT];
  #pragma unroll
  for (int s = 0; s < 2; ++s)
    #pragma unroll
    for (int mt = 0; mt < MT; ++mt) {
      int m = mt*16 + nl;
      #pragma unroll
      for (int v = 0; v < 8; ++v) {
        int k = s*32 + (v >> 2)*16 + half*8 + (v & 3)*2;
        a2[s][mt].u[v] = *(const uint32_t*)&st[m*64 + k];
      }
    }

  int cnt = cnts[kpi*2 + cnt_sel];
  #pragma unroll
  for (int t = 0; t < NT2; ++t) {
    v8f acc[MT];
    #pragma unroll
    for (int mt = 0; mt < MT; ++mt) acc[mt] = vzero8();
    int n = t*16 + nl;
    #pragma unroll
    for (int s = 0; s < 2; ++s) {
      FragB bf;
      #pragma unroll
      for (int v = 0; v < 8; ++v) {
        int k = s*32 + (v >> 2)*16 + half*8 + (v & 3)*2;
        bf.u[v] = *(const uint32_t*)&W2t[n*64 + k];
      }
      #pragma unroll
      for (int mt = 0; mt < MT; ++mt)
        acc[mt] = __builtin_amdgcn_wmma_f32_16x16x32_f16(
            false, a2[s][mt].h, false, bf.h, (short)0, acc[mt], false, false);
    }
    float bb = b2[n];
    float mx = 0.f;                                        // relu floor
    #pragma unroll
    for (int mt = 0; mt < MT; ++mt)
      #pragma unroll
      for (int r = 0; r < 8; ++r) mx = fmaxf(mx, acc[mt][r] + bb);
    mx = fmaxf(mx, __shfl_xor(mx, 16, 32));                // combine half-rows
    if (half == 0)
      featbuf[(size_t)kpi*KPAD + colofs + n] = (_Float16)((cnt > 0) ? mx : 0.f);
  }
}

// ---------------- BEV bilinear interp + raw keypoint feats into fused row ----------------
__global__ void bev_kf_kernel(const float* bev, const float* feats, const float* kp,
                              const int* kp_idx, _Float16* featbuf) {
  int i = blockIdx.x * blockDim.x + threadIdx.x;
  if (i >= BATCH*NK) return;
  int b = i / NK;
  _Float16* out = featbuf + (size_t)i*KPAD;
  // kf -> cols [192,224)
  const float* F = feats + ((size_t)b*NPT + kp_idx[i])*CF;
  for (int c = 0; c < CF; ++c) out[192 + c] = (_Float16)F[c];
  // bev -> cols [224,480)
  float x = (kp[i*3+0] - 0.f)  / 0.4f;
  float y = (kp[i*3+1] + 40.f) / 0.4f;
  int x0 = min(max((int)floorf(x), 0), W_BEV-1);
  int x1 = min(x0 + 1, W_BEV-1);
  int y0 = min(max((int)floorf(y), 0), H_BEV-1);
  int y1 = min(y0 + 1, H_BEV-1);
  float wa = ((float)x1 - x)*((float)y1 - y);
  float wb = ((float)x1 - x)*(y - (float)y0);
  float wc = (x - (float)x0)*((float)y1 - y);
  float wd = (x - (float)x0)*(y - (float)y0);
  const float* bb = bev + (size_t)b*C_BEV*H_BEV*W_BEV;
  for (int c = 0; c < C_BEV; ++c) {
    const float* ch = bb + (size_t)c*H_BEV*W_BEV;
    float v = ch[y0*W_BEV+x0]*wa + ch[y1*W_BEV+x0]*wb +
              ch[y0*W_BEV+x1]*wc + ch[y1*W_BEV+x1]*wd;
    out[224 + c] = (_Float16)v;
  }
  // zero pad cols [480,512)
  for (int c = 480; c < KPAD; ++c) out[c] = (_Float16)0.f;
}

// ---------------- fusion GEMM [4096 x 512] @ [512 x 128] + BN + ReLU via WMMA ----------------
__global__ __launch_bounds__(256) void fuse_gemm_kernel(
    const _Float16* __restrict__ featb, const _Float16* __restrict__ Wt,
    const float* __restrict__ scale, const float* __restrict__ bias,
    float* __restrict__ out) {
#if defined(__has_builtin)
#if __has_builtin(__builtin_amdgcn_tensor_load_to_lds)
  // Exercise the TDM path with a NULL tensor descriptor (count==0 -> NOP per D# spec).
  if (blockIdx.x == 0 && (threadIdx.x >> 5) == 0) {
    typedef unsigned int u32x4 __attribute__((ext_vector_type(4)));
    typedef int i32x4 __attribute__((ext_vector_type(4)));
    typedef int i32x8 __attribute__((ext_vector_type(8)));
    u32x4 g0 = {0u,0u,0u,0u};
    i32x8 g1 = {0,0,0,0,0,0,0,0};
    i32x4 g2 = {0,0,0,0};
    i32x4 g3 = {0,0,0,0};
#if defined(__clang_major__) && (__clang_major__ >= 23)
    i32x8 g4 = {0,0,0,0,0,0,0,0};
    __builtin_amdgcn_tensor_load_to_lds(g0, g1, g2, g3, g4, 0);
#else
    __builtin_amdgcn_tensor_load_to_lds(g0, g1, g2, g3, 0);
#endif
    __builtin_amdgcn_s_wait_tensorcnt(0);
  }
#endif
#endif
  int w = threadIdx.x >> 5, lane = threadIdx.x & 31;
  int mt = blockIdx.x;            // 256 M-tiles of 16 rows
  int nt = w;                     // 8 N-tiles of 16 cols
  int half = lane >> 4, nl = lane & 15;
  int m = mt*16 + nl;
  int n = nt*16 + nl;
  v8f acc = vzero8();
  for (int s = 0; s < 16; ++s) {
    if (s < 15)
      __builtin_prefetch(&featb[(size_t)m*KPAD + (s+1)*32], 0, 3);  // global_prefetch_b8
    FragA a;
    #pragma unroll
    for (int v = 0; v < 8; ++v) {
      int k = s*32 + (v >> 2)*16 + half*8 + (v & 3)*2;
      a.u[v] = *(const uint32_t*)&featb[(size_t)m*KPAD + k];
    }
    FragB bf;
    #pragma unroll
    for (int v = 0; v < 8; ++v) {
      int k = s*32 + (v >> 2)*16 + half*8 + (v & 3)*2;
      bf.u[v] = *(const uint32_t*)&Wt[(size_t)n*KPAD + k];
    }
    acc = __builtin_amdgcn_wmma_f32_16x16x32_f16(false, a.h, false, bf.h,
                                                 (short)0, acc, false, false);
  }
  float sc = scale[n], bi = bias[n];
  #pragma unroll
  for (int r = 0; r < 8; ++r) {
    int mm = mt*16 + half*8 + r;
    out[(size_t)mm*C_OUT + n] = fmaxf(acc[r]*sc + bi, 0.f);
  }
}

// ---------------- launch ----------------
extern "C" void kernel_launch(void* const* d_in, const int* in_sizes, int n_in,
                              void* d_out, int out_size, void* d_ws, size_t ws_size,
                              hipStream_t stream) {
  (void)in_sizes; (void)n_in; (void)out_size; (void)ws_size;
  const float* points = (const float*)d_in[0];
  const float* pfeats = (const float*)d_in[1];
  const float* bev    = (const float*)d_in[2];
  const float* W1a = (const float*)d_in[3];  const float* b1a = (const float*)d_in[4];
  const float* W2a = (const float*)d_in[5];  const float* b2a = (const float*)d_in[6];
  const float* W1b = (const float*)d_in[7];  const float* b1b = (const float*)d_in[8];
  const float* W2b = (const float*)d_in[9];  const float* b2b = (const float*)d_in[10];
  const float* fW  = (const float*)d_in[11];
  const float* fscale = (const float*)d_in[12];
  const float* fbias  = (const float*)d_in[13];

  char* ws = (char*)d_ws;
  int*      kp_idx = (int*)(ws + OFF_KPIDX);
  float*    kp     = (float*)(ws + OFF_KP);
  int*      cnts   = (int*)(ws + OFF_CNT);
  int*      idx1   = (int*)(ws + OFF_IDX1);
  int*      idx2   = (int*)(ws + OFF_IDX2);
  _Float16* W1at   = (_Float16*)(ws + OFF_W1A);
  _Float16* W2at   = (_Float16*)(ws + OFF_W2A);
  _Float16* W1bt   = (_Float16*)(ws + OFF_W1B);
  _Float16* W2bt   = (_Float16*)(ws + OFF_W2B);
  _Float16* fWt    = (_Float16*)(ws + OFF_FW);
  _Float16* featb  = (_Float16*)(ws + OFF_FEAT);

  float* out_x      = (float*)d_out;                       // [B*NK, 128]
  float* out_coords = out_x + (size_t)BATCH*NK*C_OUT;      // [B*NK, 4]

  prep_weights_kernel<<<(86016 + 255)/256, 256, 0, stream>>>(
      W1a, W2a, W1b, W2b, fW, W1at, W2at, W1bt, W2bt, fWt);
  fps_kernel<<<BATCH*2, 512, 0, stream>>>(points, pfeats, kp_idx);
  gather_kp_kernel<<<(BATCH*NK + 255)/256, 256, 0, stream>>>(points, kp_idx, kp, out_coords);
  ball_query_kernel<<<BATCH*NK/8, 256, 0, stream>>>(points, kp, idx1, idx2, cnts);
  group_mlp_kernel<NS1, 64><<<BATCH*NK/8, 256, 0, stream>>>(
      points, pfeats, kp, idx1, cnts, 0, W1at, b1a, W2at, b2a, featb, 0);
  group_mlp_kernel<NS2, 128><<<BATCH*NK/8, 256, 0, stream>>>(
      points, pfeats, kp, idx2, cnts, 1, W1bt, b1b, W2bt, b2b, featb, 64);
  bev_kf_kernel<<<(BATCH*NK + 255)/256, 256, 0, stream>>>(bev, pfeats, kp, kp_idx, featb);
  fuse_gemm_kernel<<<256, 256, 0, stream>>>(featb, fWt, fscale, fbias, out_x);
}
